// GCN_mamba_block_16149077033109
// MI455X (gfx1250) — matrix operand order repaired and checked by reading, not compiled, not run on hardware
//
#include <hip/hip_runtime.h>

// Problem constants (match reference)
#define NN      4096   // nodes
#define DM      256    // d_model
#define DSTATE  16
#define DTRANK  16
#define LHOPS   8
#define ALPHA_C 0.05f

// GEMM tiling
#define BM 128
#define BN 128
#define BK 32

typedef __attribute__((ext_vector_type(16))) __bf16 v16bf;
typedef __attribute__((ext_vector_type(8)))  float  v8f;

union FragU { uint4 q[2]; v16bf v; };

__device__ __forceinline__ void split_bf16(float f, __bf16& hi, __bf16& lo) {
  hi = (__bf16)f;
  lo = (__bf16)(f - (float)hi);
}

// ---- CDNA5 async global->LDS copy (ASYNCcnt path) ----
__device__ __forceinline__ void async_copy_b128(unsigned lds_byte_addr, const void* gptr) {
  asm volatile("global_load_async_to_lds_b128 %0, %1, off"
               :: "v"(lds_byte_addr),
                  "v"((unsigned long long)(uintptr_t)gptr)
               : "memory");
}
__device__ __forceinline__ void wait_async0() {
  asm volatile("s_wait_asynccnt 0" ::: "memory");
}

// Split fp32 array into bf16 hi/lo limbs (a = hi + lo, ~16 mantissa bits total)
__global__ __launch_bounds__(256)
void split_limbs(const float* __restrict__ src,
                 __bf16* __restrict__ hi, __bf16* __restrict__ lo)
{
  size_t i4 = (size_t)blockIdx.x * 256 + threadIdx.x;   // float4 index
  float4 v = ((const float4*)src)[i4];
  union { __bf16 b[4]; uint2 u; } H, L;
  split_bf16(v.x, H.b[0], L.b[0]);
  split_bf16(v.y, H.b[1], L.b[1]);
  split_bf16(v.z, H.b[2], L.b[2]);
  split_bf16(v.w, H.b[3], L.b[3]);
  ((uint2*)hi)[i4] = H.u;
  ((uint2*)lo)[i4] = L.u;
}

// GCN hop GEMM: C = scale*(A @ B) + addScale*Xadd
// A = pre-split bf16 limbs, double-buffered async global->LDS copies;
// B = fp32, split on the fly. bf16x3 WMMA accumulation in fp32.
__global__ __launch_bounds__(256)
void gemm_prop_async(const __bf16* __restrict__ Ahi, const __bf16* __restrict__ Alo, int lda,
                     const float* __restrict__ B, int ldb,
                     float* __restrict__ C, int ldc,
                     const float* __restrict__ Xadd, int ldx,
                     int M, int K, float scale, float addScale)
{
  __shared__ __align__(16) __bf16 As_hi[2][BM * BK];
  __shared__ __align__(16) __bf16 As_lo[2][BM * BK];
  __shared__ __align__(16) __bf16 Bs_hi[BN * BK];   // transposed [n][k]
  __shared__ __align__(16) __bf16 Bs_lo[BN * BK];

  const int tid  = threadIdx.x;
  const int lane = tid & 31;
  const int wid  = tid >> 5;
  const int wm   = wid >> 1;        // 0..3 -> 32-row band
  const int wn   = wid & 1;         // 0..1 -> 64-col band

  const int rowBase = blockIdx.y * BM;
  const int colBase = blockIdx.x * BN;

  const unsigned as_hi_base[2] = { (unsigned)(uintptr_t)(void*)As_hi[0],
                                   (unsigned)(uintptr_t)(void*)As_hi[1] };
  const unsigned as_lo_base[2] = { (unsigned)(uintptr_t)(void*)As_lo[0],
                                   (unsigned)(uintptr_t)(void*)As_lo[1] };

  // Issue async copy of one 128x32 bf16 limb tile into LDS buffer `buf`.
  auto issueA = [&](int kb, int buf) {
    for (int i = 0; i < 2; ++i) {
      int c  = tid + i * 256;          // 0..511 16B chunks
      int r  = c >> 2;                 // 0..127
      int kq = (c & 3) << 3;           // 0,8,16,24
      size_t goff = (size_t)(rowBase + r) * lda + kb + kq;
      async_copy_b128(as_hi_base[buf] + (unsigned)(c * 16), Ahi + goff);
      async_copy_b128(as_lo_base[buf] + (unsigned)(c * 16), Alo + goff);
    }
  };

  v8f acc[2][4];
  for (int i = 0; i < 2; ++i)
    for (int j = 0; j < 4; ++j)
      for (int v = 0; v < 8; ++v) acc[i][j][v] = 0.0f;

  const int lr = lane & 15;
  const int lh = lane >> 4;

  issueA(0, 0);
  int buf = 0;

  for (int kb = 0; kb < K; kb += BK) {
    // ---- B tile (BK x BN) fp32: split + transpose into [n][k] ----
    // (overlaps with the in-flight async A copy)
    for (int i = 0; i < 4; ++i) {
      int idx = tid + i * 256;
      int kk  = idx >> 5;                // 0..31
      int nq  = (idx & 31) << 2;         // 0..124
      const float* bp = B + (size_t)(kb + kk) * ldb + colBase + nq;
      float4 b = *(const float4*)bp;
      if (kb + BK < K) __builtin_prefetch(bp + (size_t)BK * ldb, 0, 1);
      __bf16 h, l;
      split_bf16(b.x, h, l); Bs_hi[(nq + 0) * BK + kk] = h; Bs_lo[(nq + 0) * BK + kk] = l;
      split_bf16(b.y, h, l); Bs_hi[(nq + 1) * BK + kk] = h; Bs_lo[(nq + 1) * BK + kk] = l;
      split_bf16(b.z, h, l); Bs_hi[(nq + 2) * BK + kk] = h; Bs_lo[(nq + 2) * BK + kk] = l;
      split_bf16(b.w, h, l); Bs_hi[(nq + 3) * BK + kk] = h; Bs_lo[(nq + 3) * BK + kk] = l;
    }

    wait_async0();        // A tile `buf` landed (only its asyncs outstanding)
    __syncthreads();      // publish A copy + B stores to all waves

    // Prefetch next A tile into the other buffer; overlaps the WMMAs below.
    if (kb + BK < K) issueA(kb + BK, buf ^ 1);

    // ---- fragments ----
    const __bf16* Ah = As_hi[buf];
    const __bf16* Al = As_lo[buf];
    FragU a_hi[2], a_lo[2], b_hi[4], b_lo[4];
    for (int tm = 0; tm < 2; ++tm) {
      int r = wm * 32 + tm * 16 + lr;
      const uint4* ph = (const uint4*)(Ah + r * BK);
      const uint4* pl = (const uint4*)(Al + r * BK);
      a_hi[tm].q[0] = ph[2 * lh]; a_hi[tm].q[1] = ph[2 * lh + 1];
      a_lo[tm].q[0] = pl[2 * lh]; a_lo[tm].q[1] = pl[2 * lh + 1];
    }
    for (int tn = 0; tn < 4; ++tn) {
      int nc = wn * 64 + tn * 16 + lr;
      const uint4* ph = (const uint4*)(Bs_hi + nc * BK + lh * 16);
      const uint4* pl = (const uint4*)(Bs_lo + nc * BK + lh * 16);
      b_hi[tn].q[0] = ph[0]; b_hi[tn].q[1] = ph[1];
      b_lo[tn].q[0] = pl[0]; b_lo[tn].q[1] = pl[1];
    }

    // ---- bf16x3: hi*hi + hi*lo + lo*hi ----
    for (int tm = 0; tm < 2; ++tm)
      for (int tn = 0; tn < 4; ++tn) {
        v8f c = acc[tm][tn];
        c = __builtin_amdgcn_wmma_f32_16x16x32_bf16(false, a_hi[tm].v, false, b_hi[tn].v, (short)0, c, false, false);
        c = __builtin_amdgcn_wmma_f32_16x16x32_bf16(false, a_hi[tm].v, false, b_lo[tn].v, (short)0, c, false, false);
        c = __builtin_amdgcn_wmma_f32_16x16x32_bf16(false, a_lo[tm].v, false, b_hi[tn].v, (short)0, c, false, false);
        acc[tm][tn] = c;
      }
    __syncthreads();      // all frag reads done before B (and A[buf]) are overwritten
    buf ^= 1;
  }

  for (int tm = 0; tm < 2; ++tm) {
    int rbase = rowBase + wm * 32 + tm * 16 + (lane >> 4) * 8;
    for (int tn = 0; tn < 4; ++tn) {
      int col = colBase + wn * 64 + tn * 16 + (lane & 15);
      for (int v = 0; v < 8; ++v) {
        int row = rbase + v;
        float r = scale * acc[tm][tn][v];
        r += addScale * Xadd[(size_t)row * ldx + col];
        C[(size_t)row * ldc + col] = r;
      }
    }
  }
}

// Generic fp32-input bf16x3 GEMM (used for final @ W_out)
__global__ __launch_bounds__(256)
void gemm_bf16x3(const float* __restrict__ A, int lda,
                 const float* __restrict__ B, int ldb,
                 float* __restrict__ C, int ldc,
                 int M, int K)
{
  __shared__ __align__(16) __bf16 As_hi[BM * BK];
  __shared__ __align__(16) __bf16 As_lo[BM * BK];
  __shared__ __align__(16) __bf16 Bs_hi[BN * BK];
  __shared__ __align__(16) __bf16 Bs_lo[BN * BK];

  const int tid  = threadIdx.x;
  const int lane = tid & 31;
  const int wid  = tid >> 5;
  const int wm   = wid >> 1;
  const int wn   = wid & 1;

  const int rowBase = blockIdx.y * BM;
  const int colBase = blockIdx.x * BN;

  v8f acc[2][4];
  for (int i = 0; i < 2; ++i)
    for (int j = 0; j < 4; ++j)
      for (int v = 0; v < 8; ++v) acc[i][j][v] = 0.0f;

  const int lr = lane & 15;
  const int lh = lane >> 4;

  for (int kb = 0; kb < K; kb += BK) {
    for (int i = 0; i < 4; ++i) {
      int idx = tid + i * 256;
      int r   = idx >> 3;
      int kq  = (idx & 7) << 2;
      float4 a = *(const float4*)(A + (size_t)(rowBase + r) * lda + kb + kq);
      __bf16 h, l;
      split_bf16(a.x, h, l); As_hi[r * BK + kq + 0] = h; As_lo[r * BK + kq + 0] = l;
      split_bf16(a.y, h, l); As_hi[r * BK + kq + 1] = h; As_lo[r * BK + kq + 1] = l;
      split_bf16(a.z, h, l); As_hi[r * BK + kq + 2] = h; As_lo[r * BK + kq + 2] = l;
      split_bf16(a.w, h, l); As_hi[r * BK + kq + 3] = h; As_lo[r * BK + kq + 3] = l;
    }
    for (int i = 0; i < 4; ++i) {
      int idx = tid + i * 256;
      int kk  = idx >> 5;
      int nq  = (idx & 31) << 2;
      float4 b = *(const float4*)(B + (size_t)(kb + kk) * ldb + colBase + nq);
      __bf16 h, l;
      split_bf16(b.x, h, l); Bs_hi[(nq + 0) * BK + kk] = h; Bs_lo[(nq + 0) * BK + kk] = l;
      split_bf16(b.y, h, l); Bs_hi[(nq + 1) * BK + kk] = h; Bs_lo[(nq + 1) * BK + kk] = l;
      split_bf16(b.z, h, l); Bs_hi[(nq + 2) * BK + kk] = h; Bs_lo[(nq + 2) * BK + kk] = l;
      split_bf16(b.w, h, l); Bs_hi[(nq + 3) * BK + kk] = h; Bs_lo[(nq + 3) * BK + kk] = l;
    }
    __syncthreads();

    FragU a_hi[2], a_lo[2], b_hi[4], b_lo[4];
    for (int tm = 0; tm < 2; ++tm) {
      int r = wm * 32 + tm * 16 + lr;
      const uint4* ph = (const uint4*)(As_hi + r * BK);
      const uint4* pl = (const uint4*)(As_lo + r * BK);
      a_hi[tm].q[0] = ph[2 * lh]; a_hi[tm].q[1] = ph[2 * lh + 1];
      a_lo[tm].q[0] = pl[2 * lh]; a_lo[tm].q[1] = pl[2 * lh + 1];
    }
    for (int tn = 0; tn < 4; ++tn) {
      int nc = wn * 64 + tn * 16 + lr;
      const uint4* ph = (const uint4*)(Bs_hi + nc * BK + lh * 16);
      const uint4* pl = (const uint4*)(Bs_lo + nc * BK + lh * 16);
      b_hi[tn].q[0] = ph[0]; b_hi[tn].q[1] = ph[1];
      b_lo[tn].q[0] = pl[0]; b_lo[tn].q[1] = pl[1];
    }

    for (int tm = 0; tm < 2; ++tm)
      for (int tn = 0; tn < 4; ++tn) {
        v8f c = acc[tm][tn];
        c = __builtin_amdgcn_wmma_f32_16x16x32_bf16(false, a_hi[tm].v, false, b_hi[tn].v, (short)0, c, false, false);
        c = __builtin_amdgcn_wmma_f32_16x16x32_bf16(false, a_hi[tm].v, false, b_lo[tn].v, (short)0, c, false, false);
        c = __builtin_amdgcn_wmma_f32_16x16x32_bf16(false, a_lo[tm].v, false, b_hi[tn].v, (short)0, c, false, false);
        acc[tm][tn] = c;
      }
    __syncthreads();
  }

  for (int tm = 0; tm < 2; ++tm) {
    int rbase = rowBase + wm * 32 + tm * 16 + (lane >> 4) * 8;
    for (int tn = 0; tn < 4; ++tn) {
      int col = colBase + wn * 64 + tn * 16 + (lane & 15);
      for (int v = 0; v < 8; ++v)
        C[(size_t)(rbase + v) * ldc + col] = acc[tm][tn][v];
    }
  }
}

// u[:,0,:] = x
__global__ __launch_bounds__(256)
void scatter_u0(const float* __restrict__ x, float* __restrict__ u)
{
  int g = blockIdx.x * blockDim.x + threadIdx.x;
  int n = g / DM, d = g % DM;
  u[(size_t)n * (LHOPS * DM) + d] = x[g];
}

// Per-node SSM scan
__global__ __launch_bounds__(256)
void ssm_scan(const float* __restrict__ u,
              const float* __restrict__ Wx,
              const float* __restrict__ Wdt,
              const float* __restrict__ A_log,
              const float* __restrict__ Dp,
              float* __restrict__ yb)
{
  __shared__ float u_s[LHOPS * DM];
  __shared__ float xdbl[LHOPS * 48];
  const int n = blockIdx.x;
  const int t = threadIdx.x;

  for (int i = 0; i < LHOPS; ++i)
    u_s[i * DM + t] = u[(size_t)n * (LHOPS * DM) + i * DM + t];
  __syncthreads();

  for (int idx = t; idx < LHOPS * 48; idx += 256) {
    int l = idx / 48, j = idx % 48;
    const float* ur = u_s + l * DM;
    float s = 0.f;
    for (int d = 0; d < DM; ++d) s += ur[d] * Wx[d * 48 + j];
    xdbl[idx] = fmaxf(s, 0.f);
  }
  __syncthreads();

  const int d = t;
  float a_s[DSTATE];
  for (int s = 0; s < DSTATE; ++s) a_s[s] = -__expf(A_log[d * DSTATE + s]);
  const float Dd = Dp[d];

  float state[DSTATE];
  for (int s = 0; s < DSTATE; ++s) state[s] = 0.f;

  for (int l = 0; l < LHOPS; ++l) {
    float dl = 0.f;
    for (int r = 0; r < DTRANK; ++r) dl += xdbl[l * 48 + r] * Wdt[r * DM + d];
    dl = (dl > 20.f) ? dl : log1pf(__expf(dl));
    float ul = u_s[l * DM + d];
    float du = dl * ul;
    float y = 0.f;
    for (int s = 0; s < DSTATE; ++s) {
      float dA = __expf(dl * a_s[s]);
      state[s] = dA * state[s] + du * xdbl[l * 48 + DTRANK + s];
      y += state[s] * xdbl[l * 48 + DTRANK + DSTATE + s];
    }
    yb[(size_t)n * (LHOPS * DM) + l * DM + d] = y + ul * Dd;
  }
}

extern "C" void kernel_launch(void* const* d_in, const int* in_sizes, int n_in,
                              void* d_out, int out_size, void* d_ws, size_t ws_size,
                              hipStream_t stream)
{
  (void)in_sizes; (void)n_in; (void)out_size; (void)ws_size;
  const float* x    = (const float*)d_in[0];
  const float* adj  = (const float*)d_in[1];
  const float* Wx   = (const float*)d_in[2];
  const float* Wdt  = (const float*)d_in[3];
  const float* A_lg = (const float*)d_in[4];
  const float* Dp   = (const float*)d_in[5];
  const float* Wout = (const float*)d_in[6];
  float* out = (float*)d_out;

  __bf16* adj_hi = (__bf16*)d_ws;                      // 32 MB
  __bf16* adj_lo = adj_hi + (size_t)NN * NN;           // 32 MB
  float*  u      = (float*)(adj_lo + (size_t)NN * NN); // 32 MB
  float*  yb     = u + (size_t)NN * LHOPS * DM;        // 32 MB

  // adj -> bf16 limbs (once per call; L2-resident afterwards)
  split_limbs<<<(NN * NN / 4) / 256, 256, 0, stream>>>(adj, adj_hi, adj_lo);

  scatter_u0<<<(NN * DM) / 256, 256, 0, stream>>>(x, u);

  // GCN hops: u[:,l,:] = 0.95*(adj @ u[:,l-1,:]) + 0.05*x
  dim3 g1(DM / BN, NN / BM);
  for (int l = 1; l < LHOPS; ++l) {
    gemm_prop_async<<<g1, 256, 0, stream>>>(adj_hi, adj_lo, NN,
                                            u + (size_t)(l - 1) * DM, LHOPS * DM,
                                            u + (size_t)l * DM,       LHOPS * DM,
                                            x, DM,
                                            NN, NN, 1.0f - ALPHA_C, ALPHA_C);
  }

  ssm_scan<<<NN, 256, 0, stream>>>(u, Wx, Wdt, A_lg, Dp, yb);

  // out = yb @ W_out
  dim3 g2(DM / BN, (NN * LHOPS) / BM);
  gemm_bf16x3<<<g2, 256, 0, stream>>>(yb, DM, Wout, DM, out, DM, NN * LHOPS, DM);
}
